// AdaptiveSlotAttention_19550691131589
// MI455X (gfx1250) — compile-verified, matrix operand order checked by reading, not run on hardware
//
#include <hip/hip_runtime.h>
#include <hip/hip_bf16.h>
#include <math.h>

typedef __attribute__((ext_vector_type(16))) _Float16 v16h;
typedef __attribute__((ext_vector_type(8)))  _Float16 v8h;
typedef __attribute__((ext_vector_type(8)))  float    v8f;

static constexpr int Bd   = 32;
static constexpr int Nd_  = 4096;
static constexpr int FDd  = 256;
static constexpr int SDd  = 256;
static constexpr int MSd  = 8;
static constexpr int NRd  = 4;
static constexpr int Hd   = 4;
static constexpr int Kd_  = 12;      // MS + NR slots
static constexpr int DHd  = 64;
static constexpr int ITERSd = 3;
static constexpr float EPSd = 1e-8f;

// ---------------------------------------------------------------------------
// A-fragment load from f32, unit k-stride, branch-free (mask folds row pad).
// Fragment k-pattern (CDNA5 16-bit 16x32): elements 0..7 at off, 8..15 at
// off+16, where off = k0 + (lane>=16)*8.
// ---------------------------------------------------------------------------
__device__ inline v16h ld_fragA(const float* rowPtr, int off, float msk) {
    const float4* b0 = (const float4*)(rowPtr + off);
    float4 x0 = b0[0];
    float4 x1 = b0[1];
    float4 x2 = b0[4];   // +16 floats
    float4 x3 = b0[5];
    float t[16] = { x0.x, x0.y, x0.z, x0.w,  x1.x, x1.y, x1.z, x1.w,
                    x2.x, x2.y, x2.z, x2.w,  x3.x, x3.y, x3.z, x3.w };
    v16h o;
#pragma unroll
    for (int e = 0; e < 16; ++e) o[e] = (_Float16)(t[e] * msk);
    return o;
}

// B-fragment load straight from f16 memory: two 16-byte loads, no converts.
__device__ inline v16h ld_fragB(const _Float16* rowPtr, int off) {
    v8h a = *(const v8h*)(rowPtr + off);
    v8h b = *(const v8h*)(rowPtr + off + 16);
    v16h o;
#pragma unroll
    for (int e = 0; e < 8; ++e) { o[e] = a[e]; o[8 + e] = b[e]; }
    return o;
}

__device__ inline void store_tile(void* C, size_t cBase, const float* cab,
                                  const float* bias, v8f acc, int m0, int nb,
                                  int lane, int hi, int M, int Cld, int CaLd,
                                  float alpha, int act, int storeT, int storeH) {
    int n = nb + (lane & 15);
    float bv = bias ? bias[n] : 0.0f;
#pragma unroll
    for (int r = 0; r < 8; ++r) {
        int m = m0 + r + hi * 8;
        if (m < M) {
            float v = acc[r] * alpha + bv;
            if (act == 1)      v = fmaxf(v, 0.0f);
            else if (act == 2) v = 0.5f * v * (1.0f + erff(v * 0.70710678118f));
            if (cab) v += cab[(size_t)m * CaLd + n];
            size_t idx = cBase + (storeT ? ((size_t)n * Cld + m)
                                         : ((size_t)m * Cld + n));
            if (storeH) ((_Float16*)C)[idx] = (_Float16)v;
            else        ((float*)C)[idx]    = v;
        }
    }
}

// ---------------------------------------------------------------------------
// Batched strided GEMM: C = act(alpha * A @ B + bias) + Cadd.
// A (M x K) f32 unit k-stride; B given as B^T (Ndim x K), f16, unit k-stride.
// One wave computes a 16x64 macro-tile: one A fragment feeds four WMMAs.
// Optional transposed store (storeT) and f16 store (storeH).
// ---------------------------------------------------------------------------
__global__ __launch_bounds__(256)
void wmma_gemm(const float* A,     long long sA1, long long sA2, int Ars,
               const _Float16* Bt, long long sB1, long long sB2, int Brs,
               void* C,            long long sC1, long long sC2, int Cld,
               const float* bias,
               const float* rowScale, long long sRS1,
               const float* Cadd, long long sCa1, long long sCa2, int CaLd,
               int M, int Ndim, int Kdim, float alpha, int act,
               int storeT, int storeH) {
    int wav = blockIdx.x * (blockDim.x >> 5) + (threadIdx.x >> 5);
    int tm  = (M + 15) >> 4;
    int tn4 = Ndim >> 6;
    if (wav >= tm * tn4) return;           // wave-uniform: EXEC all-1 for WMMA
    long long b1 = blockIdx.y, b2 = blockIdx.z;
    const float*    Ab = A  + b1 * sA1 + b2 * sA2;
    const _Float16* Bb = Bt + b1 * sB1 + b2 * sB2;
    size_t cBase = (size_t)(b1 * sC1 + b2 * sC2);
    const float* rsb = rowScale ? rowScale + b1 * sRS1 : nullptr;
    const float* cab = Cadd ? Cadd + b1 * sCa1 + b2 * sCa2 : nullptr;

    int ti = wav / tn4, tj = wav % tn4;
    int m0 = ti * 16, n0 = tj * 64;
    int lane = threadIdx.x & 31;
    int hi   = lane >> 4;

    int  rA   = m0 + (lane & 15);
    bool arok = (rA < M);
    float amsk = arok ? (rsb ? rsb[rA] : 1.0f) : 0.0f;
    int  rAe  = arok ? rA : (M - 1);
    const float*    aRow  = Ab + (size_t)rAe * Ars;
    const _Float16* bRow0 = Bb + (size_t)(n0 + (lane & 15)) * Brs;
    const _Float16* bRow1 = bRow0 + (size_t)16 * Brs;
    const _Float16* bRow2 = bRow0 + (size_t)32 * Brs;
    const _Float16* bRow3 = bRow0 + (size_t)48 * Brs;

    v8f acc0 = {}, acc1 = {}, acc2 = {}, acc3 = {};
#pragma unroll 2
    for (int k0 = 0; k0 < Kdim; k0 += 32) {
        int off = k0 + hi * 8;
        v16h af = ld_fragA(aRow, off, amsk);
        v16h b0 = ld_fragB(bRow0, off);
        v16h b1f = ld_fragB(bRow1, off);
        v16h b2f = ld_fragB(bRow2, off);
        v16h b3f = ld_fragB(bRow3, off);
        acc0 = __builtin_amdgcn_wmma_f32_16x16x32_f16(
                   false, af, false, b0,  (short)0, acc0, false, false);
        acc1 = __builtin_amdgcn_wmma_f32_16x16x32_f16(
                   false, af, false, b1f, (short)0, acc1, false, false);
        acc2 = __builtin_amdgcn_wmma_f32_16x16x32_f16(
                   false, af, false, b2f, (short)0, acc2, false, false);
        acc3 = __builtin_amdgcn_wmma_f32_16x16x32_f16(
                   false, af, false, b3f, (short)0, acc3, false, false);
    }
    store_tile(C, cBase, cab, bias, acc0, m0, n0,      lane, hi, M, Cld, CaLd, alpha, act, storeT, storeH);
    store_tile(C, cBase, cab, bias, acc1, m0, n0 + 16, lane, hi, M, Cld, CaLd, alpha, act, storeT, storeH);
    store_tile(C, cBase, cab, bias, acc2, m0, n0 + 32, lane, hi, M, Cld, CaLd, alpha, act, storeT, storeH);
    store_tile(C, cBase, cab, bias, acc3, m0, n0 + 48, lane, hi, M, Cld, CaLd, alpha, act, storeT, storeH);
}

// wt[n*K + k] = (f16) w[k*N + n]
__global__ void transpose_h(const float* w, _Float16* wt, int Kk, int Nn) {
    int i = blockIdx.x * blockDim.x + threadIdx.x;
    if (i >= Kk * Nn) return;
    int n = i / Kk, k = i % Kk;
    wt[i] = (_Float16)w[k * Nn + n];
}

// bm = sigmoid(h1 . w2 + b2), one thread per token
__global__ void bm_kernel(const float* h1, const float* w2, const float* b2,
                          float* bm, int T) {
    int t = blockIdx.x * blockDim.x + threadIdx.x;
    if (t >= T) return;
    const float* h = h1 + (size_t)t * 64;
    float acc = b2[0];
#pragma unroll 8
    for (int j = 0; j < 64; ++j) acc += h[j] * w2[j];
    bm[t] = 1.0f / (1.0f + expf(-acc));
}

// LayerNorm over D, one wave per row
__global__ void ln_kernel(const float* x, float* y, const float* g, const float* b,
                          int rows, int D) {
    int row = blockIdx.x * (blockDim.x >> 5) + (threadIdx.x >> 5);
    if (row >= rows) return;
    int lane = threadIdx.x & 31;
    const float* xr = x + (size_t)row * D;
    float s = 0.f, s2 = 0.f;
    for (int i = lane; i < D; i += 32) { float v = xr[i]; s += v; s2 += v * v; }
    for (int o = 16; o > 0; o >>= 1) { s += __shfl_xor(s, o, 32); s2 += __shfl_xor(s2, o, 32); }
    float mean = s / D;
    float var  = s2 / D - mean * mean;
    float inv  = rsqrtf(var + 1e-5f);
    float* yr = y + (size_t)row * D;
    for (int i = lane; i < D; i += 32) yr[i] = (xr[i] - mean) * inv * g[i] + b[i];
}

// s = mu + exp(ls) * noise  (mu/ls broadcast over batch)
__global__ void s_init(float* s, const float* mu, const float* ls,
                       const float* noise, int total, int perB) {
    int i = blockIdx.x * blockDim.x + threadIdx.x;
    if (i >= total) return;
    int j = i % perB;
    s[i] = mu[j] + expf(ls[j]) * noise[i];
}

// softmax across the 48 (k,h) slot-heads for each (b,n); optional ao output
__global__ void slot_softmax(float* d, float* ao, int writeAo) {
    int t = blockIdx.x * blockDim.x + threadIdx.x;
    if (t >= Bd * Nd_) return;
    int b = t / Nd_, n = t % Nd_;
    float vals[Kd_ * Hd];
    size_t base = ((size_t)b * Kd_ * Hd) * Nd_ + n;
    float mx = -1e30f;
#pragma unroll
    for (int i = 0; i < Kd_ * Hd; ++i) {
        vals[i] = d[base + (size_t)i * Nd_];
        mx = fmaxf(mx, vals[i]);
    }
    float sum = 0.f;
#pragma unroll
    for (int i = 0; i < Kd_ * Hd; ++i) { vals[i] = expf(vals[i] - mx); sum += vals[i]; }
    float inv = 1.0f / sum;
#pragma unroll
    for (int i = 0; i < Kd_ * Hd; ++i) {
        vals[i] *= inv;
        d[base + (size_t)i * Nd_] = vals[i];
    }
    if (writeAo) {
#pragma unroll
        for (int k = 0; k < Kd_; ++k) {
            float m = 0.f;
#pragma unroll
            for (int h = 0; h < Hd; ++h) m += vals[k * Hd + h];
            ao[((size_t)b * Kd_ + k) * Nd_ + n] = m * (1.0f / Hd);
        }
    }
}

// per-row sum over len (row-contiguous)
__global__ void rowsum_kernel(const float* a, float* S, int len) {
    __shared__ float red[256];
    size_t r = blockIdx.x;
    float s = 0.f;
    for (int i = threadIdx.x; i < len; i += blockDim.x) s += a[r * len + i];
    red[threadIdx.x] = s;
    __syncthreads();
    for (int o = 128; o > 0; o >>= 1) {
        if ((int)threadIdx.x < o) red[threadIdx.x] += red[threadIdx.x + o];
        __syncthreads();
    }
    if (threadIdx.x == 0) S[r] = red[0];
}

// a = (a + EPS) / (S[row] + EPS)
__global__ void renorm_kernel(float* a, const float* S, long long total, int len) {
    long long i = (long long)blockIdx.x * blockDim.x + threadIdx.x;
    if (i >= total) return;
    a[i] = (a[i] + EPSd) / (S[i / len] + EPSd);
}

// GRU gate combine, in-place update of s
__global__ void gru_combine(const float* gi, const float* gh, float* s, int total) {
    int i = blockIdx.x * blockDim.x + threadIdx.x;
    if (i >= total) return;
    int r = i / SDd, c = i % SDd;
    size_t o = (size_t)r * 3 * SDd + c;
    float rg = 1.0f / (1.0f + expf(-(gi[o] + gh[o])));
    float z  = 1.0f / (1.0f + expf(-(gi[o + SDd] + gh[o + SDd])));
    float nn = tanhf(gi[o + 2 * SDd] + rg * gh[o + 2 * SDd]);
    float hp = s[i];
    s[i] = (1.0f - z) * nn + z * hp;
}

// Final gating head: gs MLP, decisions, kp, obj*dec and reg outputs
__global__ void head_kernel(const float* s, const float* w1, const float* b1,
                            const float* w2, const float* b2,
                            float* out_obj, float* out_reg,
                            float* out_dec, float* out_kp) {
    int b    = blockIdx.x;
    int warp = threadIdx.x >> 5;    // warp == slot k (8 warps, MS=8)
    int lane = threadIdx.x & 31;
    __shared__ float lo[MSd][2];
    __shared__ float decs[MSd];

    const float* obj = s + (size_t)(b * Kd_ + warp) * SDd;
    float a0 = 0.f, a1 = 0.f;
    for (int j = lane; j < 128; j += 32) {
        float hj = b1[j];
        for (int c = 0; c < SDd; ++c) hj += obj[c] * w1[c * 128 + j];
        hj = fmaxf(hj, 0.0f);
        a0 += hj * w2[j * 2];
        a1 += hj * w2[j * 2 + 1];
    }
    for (int o = 16; o > 0; o >>= 1) {
        a0 += __shfl_xor(a0, o, 32);
        a1 += __shfl_xor(a1, o, 32);
    }
    if (lane == 0) { lo[warp][0] = a0 + b2[0]; lo[warp][1] = a1 + b2[1]; }
    __syncthreads();

    if (threadIdx.x == 0) {
        float cnt = 0.f; int best = 0; float bestv = lo[0][1];
        for (int k = 0; k < MSd; ++k) {
            float dk = (lo[k][1] > lo[k][0]) ? 1.0f : 0.0f;   // argmax==1
            decs[k] = dk; cnt += dk;
            if (lo[k][1] > bestv) { bestv = lo[k][1]; best = k; }
        }
        if (cnt < 1.0f)
            for (int k = 0; k < MSd; ++k) decs[k] = (k == best) ? 1.0f : 0.0f;
    }
    __syncthreads();

    if ((int)threadIdx.x < MSd) {
        int k = threadIdx.x;
        out_dec[b * MSd + k] = decs[k];
        float l0 = lo[k][0], l1 = lo[k][1];
        float m  = fmaxf(l0, l1);
        float e0 = expf(l0 - m), e1 = expf(l1 - m);
        out_kp[b * MSd + k] = e1 / (e0 + e1);
    }
    for (int i = threadIdx.x; i < MSd * SDd; i += blockDim.x) {
        int k = i / SDd, c = i % SDd;
        out_obj[(size_t)b * MSd * SDd + i] = s[(size_t)(b * Kd_ + k) * SDd + c] * decs[k];
    }
    for (int i = threadIdx.x; i < NRd * SDd; i += blockDim.x) {
        int k = i / SDd, c = i % SDd;
        out_reg[(size_t)b * NRd * SDd + i] = s[(size_t)(b * Kd_ + MSd + k) * SDd + c];
    }
}

// ---------------------------------------------------------------------------
extern "C" void kernel_launch(void* const* d_in, const int* in_sizes, int n_in,
                              void* d_out, int out_size, void* d_ws, size_t ws_size,
                              hipStream_t stream) {
    const float* f      = (const float*)d_in[0];
    const float* noise  = (const float*)d_in[1];
    const float* bgm_w1 = (const float*)d_in[2];
    const float* bgm_b1 = (const float*)d_in[3];
    const float* bgm_w2 = (const float*)d_in[4];
    const float* bgm_b2 = (const float*)d_in[5];
    const float* ip_w   = (const float*)d_in[6];
    const float* ip_b   = (const float*)d_in[7];
    const float* mu     = (const float*)d_in[8];
    const float* ls     = (const float*)d_in[9];
    const float* ni_g   = (const float*)d_in[10];
    const float* ni_b   = (const float*)d_in[11];
    const float* ns_g   = (const float*)d_in[12];
    const float* ns_b   = (const float*)d_in[13];
    const float* q_w    = (const float*)d_in[14];
    const float* k_w    = (const float*)d_in[15];
    const float* v_w    = (const float*)d_in[16];
    const float* gru_wi = (const float*)d_in[17];
    const float* gru_wh = (const float*)d_in[18];
    const float* gru_bi = (const float*)d_in[19];
    const float* gru_bh = (const float*)d_in[20];
    const float* ff_g   = (const float*)d_in[21];
    const float* ff_b   = (const float*)d_in[22];
    const float* ff_w1  = (const float*)d_in[23];
    const float* ff_b1  = (const float*)d_in[24];
    const float* ff_w2  = (const float*)d_in[25];
    const float* ff_b2  = (const float*)d_in[26];
    const float* gs_w1  = (const float*)d_in[27];
    const float* gs_b1  = (const float*)d_in[28];
    const float* gs_w2  = (const float*)d_in[29];
    const float* gs_b2  = (const float*)d_in[30];
    (void)in_sizes; (void)n_in; (void)out_size; (void)ws_size;

    float* out = (float*)d_out;
    const int T = Bd * Nd_;                              // 131072 tokens
    float* out_obj = out;                                //  65536
    float* out_reg = out + 65536;                        //  32768
    float* out_bm  = out + 98304;                        // 131072
    float* out_ao  = out + 229376;                       // 1572864
    float* out_dec = out + 1802240;                      //    256
    float* out_kp  = out + 1802496;                      //    256

    // workspace layout (element units noted per region)
    float* ws   = (float*)d_ws;
    float* reg0 = ws;                                    // 33.5M f32 (tmpX, later slot pool)
    float* inp  = ws + 33554432;                         // 33.5M f32
    float* kkf  = inp + 33554432;                        // region: h1 (f32) early, then kk (f16)
    _Float16* kkb = (_Float16*)kkf;                      // kk f16 [b,n,h*64+dh]
    float* vvf  = kkf + 33554432;                        // region for vvT (f16)
    _Float16* vvT = (_Float16*)vvf;                      // vvT f16 [b, h*64+dh, n]
    _Float16* wT  = (_Float16*)(vvf + 33554432);         // transposed weights, f16
    _Float16* t_bgm = wT;                                // 64x256
    _Float16* t_ip  = t_bgm + 64 * 256;                  // 256x256
    _Float16* t_k   = t_ip  + 256 * 256;
    _Float16* t_v   = t_k   + 256 * 256;
    _Float16* t_q   = t_v   + 256 * 256;
    _Float16* t_wi  = t_q   + 256 * 256;                 // 768x256
    _Float16* t_wh  = t_wi  + 768 * 256;
    _Float16* t_ff1 = t_wh  + 768 * 256;                 // 1024x256
    _Float16* t_ff2 = t_ff1 + 1024 * 256;                // 256x1024
    // slot pool inside region0 (alive only after tmpX is dead)
    float* attA = reg0;                                  // B*K*H*N = 6291456
    float* sS   = reg0 + 6291456;                        // 98304
    float* sln  = sS   + 98304;
    float* qq   = sln  + 98304;
    float* uu   = qq   + 98304;
    float* gi   = uu   + 98304;                          // 294912
    float* gh   = gi   + 294912;
    float* yy   = gh   + 294912;                         // 98304
    float* ffh  = yy   + 98304;                          // 393216
    float* Ssum = ffh  + 393216;                         // 1536

    auto tr = [&](const float* w, _Float16* wt, int Kk, int Nn) {
        transpose_h<<<(Kk * Nn + 255) / 256, 256, 0, stream>>>(w, wt, Kk, Nn);
    };
    auto gemm = [&](const float* A, long long sA1, long long sA2, int Ars,
                    const _Float16* Bt, long long sB1, long long sB2, int Brs,
                    void* C, long long sC1, long long sC2, int Cld,
                    const float* bias, const float* rsc, long long sRS1,
                    const float* cadd, long long sCa1, long long sCa2, int CaLd,
                    int M, int Nn, int Kk, float alpha, int act,
                    int batch1, int batch2, int storeT, int storeH) {
        int tiles = ((M + 15) / 16) * (Nn / 64);
        dim3 grid((tiles + 7) / 8, batch1, batch2);
        wmma_gemm<<<grid, 256, 0, stream>>>(A, sA1, sA2, Ars, Bt, sB1, sB2, Brs,
                                            C, sC1, sC2, Cld, bias, rsc, sRS1,
                                            cadd, sCa1, sCa2, CaLd,
                                            M, Nn, Kk, alpha, act, storeT, storeH);
    };

    // ---- transpose all weights once (f16, unit k-stride B operands)
    tr(bgm_w1, t_bgm, 256, 64);
    tr(ip_w,   t_ip,  256, 256);
    tr(k_w,    t_k,   256, 256);
    tr(v_w,    t_v,   256, 256);
    tr(q_w,    t_q,   256, 256);
    tr(gru_wi, t_wi,  256, 768);
    tr(gru_wh, t_wh,  256, 768);
    tr(ff_w1,  t_ff1, 256, 1024);
    tr(ff_w2,  t_ff2, 1024, 256);

    // ---- Phase 1: gate  h1 = relu(f @ bgm_w1 + b1);  bm = sigmoid(h1.w2 + b2)
    float* h1 = kkf;   // f32, reuses kk region (kk written later)
    gemm(f, 0, 0, FDd,  t_bgm, 0, 0, FDd,  h1, 0, 0, 64,
         bgm_b1, nullptr, 0, nullptr, 0, 0, 0,  T, 64, FDd, 1.0f, 1, 1, 1, 0, 0);
    bm_kernel<<<(T + 255) / 256, 256, 0, stream>>>(h1, bgm_w2, bgm_b2, out_bm, T);

    // ---- Phase 2: x = (f * bm) @ ip_w + ip_b ; inp = LN(x)
    float* tmpX = reg0;
    gemm(f, 0, 0, FDd,  t_ip, 0, 0, FDd,  tmpX, 0, 0, SDd,
         ip_b, out_bm, 0, nullptr, 0, 0, 0,  T, SDd, FDd, 1.0f, 0, 1, 1, 0, 0);
    ln_kernel<<<(T + 7) / 8, 256, 0, stream>>>(tmpX, inp, ni_g, ni_b, T, SDd);

    // ---- Phase 3: kk = inp @ k_w (f16) ; vvT[b,c,n] = (inp @ v_w)[b,n,c] (f16)
    gemm(inp, 0, 0, SDd,  t_k, 0, 0, SDd,  kkb, 0, 0, SDd,
         nullptr, nullptr, 0, nullptr, 0, 0, 0,  T, SDd, SDd, 1.0f, 0, 1, 1, 0, 1);
    gemm(inp, (long long)Nd_ * SDd, 0, SDd,  t_v, 0, 0, SDd,
         vvT, (long long)SDd * Nd_, 0, Nd_,
         nullptr, nullptr, 0, nullptr, 0, 0, 0,  Nd_, SDd, SDd, 1.0f, 0, Bd, 1, 1, 1);

    // ---- slots init: s = mu + exp(ls) * noise
    const int SK = Bd * Kd_ * SDd;
    s_init<<<(SK + 255) / 256, 256, 0, stream>>>(sS, mu, ls, noise, SK, Kd_ * SDd);

    const float scale = 0.125f; // DH^-0.5
    for (int it = 0; it < ITERSd; ++it) {
        // q = LN(s) @ q_w
        ln_kernel<<<(Bd * Kd_ + 7) / 8, 256, 0, stream>>>(sS, sln, ns_g, ns_b, Bd * Kd_, SDd);
        gemm(sln, 0, 0, SDd,  t_q, 0, 0, SDd,  qq, 0, 0, SDd,
             nullptr, nullptr, 0, nullptr, 0, 0, 0,  Bd * Kd_, SDd, SDd, 1.0f, 0, 1, 1, 0, 0);

        // d[b,k,h,n] = scale * q[b,k,h,:] . kk[b,n,h,:]   (batch: b, h)
        gemm(qq,  (long long)Kd_ * SDd, DHd, SDd,
             kkb, (long long)Nd_ * SDd, DHd, SDd,
             attA, (long long)Kd_ * Hd * Nd_, Nd_, Hd * Nd_,
             nullptr, nullptr, 0, nullptr, 0, 0, 0,
             Kd_, Nd_, DHd, scale, 0, Bd, Hd, 0, 0);

        // softmax across 48 slot-heads per (b,n); write ao on last iter
        slot_softmax<<<(T + 255) / 256, 256, 0, stream>>>(attA, out_ao, it == ITERSd - 1);

        // a = (a + EPS) / (sum_n a + EPS)
        rowsum_kernel<<<Bd * Kd_ * Hd, 256, 0, stream>>>(attA, Ssum, Nd_);
        long long tot = (long long)Bd * Kd_ * Hd * Nd_;
        renorm_kernel<<<(unsigned)((tot + 255) / 256), 256, 0, stream>>>(attA, Ssum, tot, Nd_);

        // u[b,k,h,:] = sum_n a[b,k,h,n] * vvT[b, h*64+dh, n]   (batch: b, h)
        gemm(attA, (long long)Kd_ * Hd * Nd_, Nd_, Hd * Nd_,
             vvT,  (long long)SDd * Nd_, (long long)DHd * Nd_, Nd_,
             uu,   (long long)Kd_ * SDd, DHd, SDd,
             nullptr, nullptr, 0, nullptr, 0, 0, 0,
             Kd_, DHd, Nd_, 1.0f, 0, Bd, Hd, 0, 0);

        // GRU: gi = u @ gru_wi + bi ; gh = s @ gru_wh + bh ; combine
        gemm(uu, 0, 0, SDd,  t_wi, 0, 0, SDd,  gi, 0, 0, 3 * SDd,
             gru_bi, nullptr, 0, nullptr, 0, 0, 0,  Bd * Kd_, 3 * SDd, SDd, 1.0f, 0, 1, 1, 0, 0);
        gemm(sS, 0, 0, SDd,  t_wh, 0, 0, SDd,  gh, 0, 0, 3 * SDd,
             gru_bh, nullptr, 0, nullptr, 0, 0, 0,  Bd * Kd_, 3 * SDd, SDd, 1.0f, 0, 1, 1, 0, 0);
        gru_combine<<<(SK + 255) / 256, 256, 0, stream>>>(gi, gh, sS, SK);

        // FF residual: s += gelu(LN(s) @ ff_w1 + b1) @ ff_w2 + b2
        ln_kernel<<<(Bd * Kd_ + 7) / 8, 256, 0, stream>>>(sS, yy, ff_g, ff_b, Bd * Kd_, SDd);
        gemm(yy, 0, 0, SDd,  t_ff1, 0, 0, SDd,  ffh, 0, 0, 4 * SDd,
             ff_b1, nullptr, 0, nullptr, 0, 0, 0,  Bd * Kd_, 4 * SDd, SDd, 1.0f, 2, 1, 1, 0, 0);
        gemm(ffh, 0, 0, 4 * SDd,  t_ff2, 0, 0, 4 * SDd,  sS, 0, 0, SDd,
             ff_b2, nullptr, 0, sS, 0, 0, SDd,  Bd * Kd_, SDd, 4 * SDd, 1.0f, 0, 1, 1, 0, 0);
    }

    // ---- final gating head, decisions, outputs
    head_kernel<<<Bd, 256, 0, stream>>>(sS, gs_w1, gs_b1, gs_w2, gs_b2,
                                        out_obj, out_reg, out_dec, out_kp);
}
